// HamiltonianBlockGenLayer_70162585747873
// MI455X (gfx1250) — matrix operand order repaired and checked by reading, not compiled, not run on hardware
//
#include <hip/hip_runtime.h>
#include <hip/hip_bf16.h>

typedef __bf16 bf16;
typedef bf16  v16bf __attribute__((ext_vector_type(16)));
typedef bf16  v8bf  __attribute__((ext_vector_type(8)));
typedef float v8f   __attribute__((ext_vector_type(8)));

#define NATOMS 800
#define FDIM   64
#define BDIM   14
#define BB     196          // BDIM*BDIM
#define HROW   (NATOMS*BDIM) // 11200
#define NPAIRS 319600       // N*(N-1)/2

// ---- workspace layout (bf16 elements): weight fragments ----
// fragment = 512 bf16 (32 lanes x 16 elems), B-matrix layout for wmma_f32_16x16x32_bf16
#define WO1_KT 6
#define WO1_NT 8
#define WO2_KT 4
#define WO2_NT 13
#define WD1_KT 4
#define WD1_NT 8
#define WD2_KT 4
#define WD2_NT 13
#define WO1_OFF 0
#define WO2_OFF (WO1_OFF + WO1_KT*WO1_NT*512)
#define WD1_OFF (WO2_OFF + WO2_KT*WO2_NT*512)
#define WD2_OFF (WD1_OFF + WD1_KT*WD1_NT*512)

// Repack a row-major f32 weight [K x Ncols] into bf16 WMMA B-fragments.
// Fragment (kt,nt), lane L (n = L%16, khalf = L/16), elem e: K = kt*32 + khalf*8 + (e&7) + (e>>3)*16,
// N = nt*16 + n. Out-of-range N (196..207) zero-filled.
__global__ void pack_weights_kernel(const float* __restrict__ W, bf16* __restrict__ out,
                                    int K, int Ncols, int NT, int total) {
    int t = blockIdx.x * blockDim.x + threadIdx.x;
    if (t >= total) return;
    int frag = t >> 9;
    int within = t & 511;
    int lane = within >> 4;
    int e = within & 15;
    int kt = frag / NT;
    int nt = frag - kt * NT;
    int k = kt * 32 + (lane >> 4) * 8 + (e & 7) + (e >> 3) * 16;
    int n = nt * 16 + (lane & 15);
    float v = (k < K && n < Ncols) ? W[(size_t)k * Ncols + n] : 0.0f;
    out[t] = (bf16)v;
}

__device__ __forceinline__ v16bf make_a_frag(const bf16* row, int kbase) {
    // A layout (16-bit, 16x32): elems 0..7 -> K = kbase..kbase+7, elems 8..15 -> K+16..K+23
    v8bf lo = *(const v8bf*)(row + kbase);
    v8bf hi = *(const v8bf*)(row + kbase + 16);
    return __builtin_shufflevector(lo, hi, 0,1,2,3,4,5,6,7,8,9,10,11,12,13,14,15);
}

__device__ __forceinline__ v16bf load_b_frag(const bf16* __restrict__ Wf, int frag, int lane) {
    return *(const v16bf*)(Wf + ((size_t)frag * 32 + lane) * 16);
}

// ------------------- off-diagonal pair blocks -------------------
// One block = 16 pairs, 128 threads (4 waves). LDS-staged bf16 activations.
__global__ void __launch_bounds__(128)
pair_kernel(const float* __restrict__ nodes, const float* __restrict__ edges,
            const float* __restrict__ atom_in_block,
            const bf16* __restrict__ Wo1f, const float* __restrict__ bo1,
            const bf16* __restrict__ Wo2f, const float* __restrict__ bo2,
            const int* __restrict__ pair_i, const int* __restrict__ pair_j,
            float* __restrict__ H) {
    __shared__ __align__(16) bf16 Xs[16 * 192];   // [pair][3F]
    __shared__ __align__(16) bf16 H1s[16 * 128];  // [pair][HID]
    __shared__ int pis[16], pjs[16];

    const int tid  = threadIdx.x;
    const int lane = tid & 31;
    const int wv   = tid >> 5;
    const int h    = lane >> 4;   // K-half select
    const int nl   = lane & 15;   // N within tile (B/C/D); also M for A
    const int p0   = blockIdx.x * 16;

    if (tid < 16) { pis[tid] = pair_i[p0 + tid]; pjs[tid] = pair_j[p0 + tid]; }
    __syncthreads();

    // stage xo = concat(node_i, node_j, edge_ij) as bf16
    for (int t = tid; t < 16 * 192; t += 128) {
        int m = t / 192, f = t - m * 192;
        int i = pis[m], j = pjs[m];
        float v;
        if (f < 64)       v = nodes[(size_t)i * FDIM + f];
        else if (f < 128) v = nodes[(size_t)j * FDIM + (f - 64)];
        else              v = edges[((size_t)i * NATOMS + j) * FDIM + (f - 128)];
        Xs[t] = (bf16)v;
    }
    __syncthreads();

    // ---- layer 1: [16x192] @ [192x128], relu -> LDS bf16 ----
    const bf16* arow1 = Xs + nl * 192;
    #pragma unroll
    for (int q = 0; q < 2; ++q) {
        int nt = 2 * wv + q;
        v8f acc = {};
        #pragma unroll
        for (int kt = 0; kt < WO1_KT; ++kt) {
            v16bf a = make_a_frag(arow1, kt * 32 + h * 8);
            v16bf b = load_b_frag(Wo1f, kt * WO1_NT + nt, lane);
            acc = __builtin_amdgcn_wmma_f32_16x16x32_bf16(false, a, false, b,
                                                          (short)0, acc, false, false);
        }
        float bias = bo1[nt * 16 + nl];
        #pragma unroll
        for (int r = 0; r < 8; ++r) {
            float v = acc[r] + bias;
            v = v > 0.0f ? v : 0.0f;
            H1s[(r + 8 * h) * 128 + nt * 16 + nl] = (bf16)v;
        }
    }
    __syncthreads();

    // ---- layer 2: [16x128] @ [128x196(+pad)] + bias + residual, symmetric scatter ----
    const bf16* arow2 = H1s + nl * 128;
    for (int nt = wv; nt < WO2_NT; nt += 4) {
        v8f acc = {};
        #pragma unroll
        for (int kt = 0; kt < WO2_KT; ++kt) {
            v16bf a = make_a_frag(arow2, kt * 32 + h * 8);
            v16bf b = load_b_frag(Wo2f, kt * WO2_NT + nt, lane);
            acc = __builtin_amdgcn_wmma_f32_16x16x32_bf16(false, a, false, b,
                                                          (short)0, acc, false, false);
        }
        int col = nt * 16 + nl;
        if (col < BB) {
            float bias = bo2[col];
            int r2 = col / BDIM, c2 = col - r2 * BDIM;
            #pragma unroll
            for (int r = 0; r < 8; ++r) {
                int m = r + 8 * h;
                int i = pis[m], j = pjs[m];
                float v = acc[r] + bias + atom_in_block[(size_t)(p0 + m) * BB + col];
                H[(size_t)(i * BDIM + r2) * HROW + (j * BDIM + c2)] = v;
                H[(size_t)(j * BDIM + c2) * HROW + (i * BDIM + r2)] = v;
            }
        }
    }
}

// ------------------- diagonal blocks -------------------
__global__ void __launch_bounds__(128)
diag_kernel(const float* __restrict__ nodes, const float* __restrict__ edges,
            const float* __restrict__ atom_block,
            const bf16* __restrict__ Wd1f, const float* __restrict__ bd1,
            const bf16* __restrict__ Wd2f, const float* __restrict__ bd2,
            float* __restrict__ H) {
    __shared__ __align__(16) bf16 Xs[16 * 128];
    __shared__ __align__(16) bf16 H1s[16 * 128];

    const int tid  = threadIdx.x;
    const int lane = tid & 31;
    const int wv   = tid >> 5;
    const int h    = lane >> 4;
    const int nl   = lane & 15;
    const int a0   = blockIdx.x * 16;

    // stage xd = concat(node_i, edge_ii) as bf16
    for (int t = tid; t < 16 * 128; t += 128) {
        int m = t / 128, f = t - m * 128;
        int i = a0 + m;
        float v = (f < 64) ? nodes[(size_t)i * FDIM + f]
                           : edges[((size_t)i * NATOMS + i) * FDIM + (f - 64)];
        Xs[t] = (bf16)v;
    }
    __syncthreads();

    const bf16* arow1 = Xs + nl * 128;
    #pragma unroll
    for (int q = 0; q < 2; ++q) {
        int nt = 2 * wv + q;
        v8f acc = {};
        #pragma unroll
        for (int kt = 0; kt < WD1_KT; ++kt) {
            v16bf a = make_a_frag(arow1, kt * 32 + h * 8);
            v16bf b = load_b_frag(Wd1f, kt * WD1_NT + nt, lane);
            acc = __builtin_amdgcn_wmma_f32_16x16x32_bf16(false, a, false, b,
                                                          (short)0, acc, false, false);
        }
        float bias = bd1[nt * 16 + nl];
        #pragma unroll
        for (int r = 0; r < 8; ++r) {
            float v = acc[r] + bias;
            v = v > 0.0f ? v : 0.0f;
            H1s[(r + 8 * h) * 128 + nt * 16 + nl] = (bf16)v;
        }
    }
    __syncthreads();

    const bf16* arow2 = H1s + nl * 128;
    for (int nt = wv; nt < WD2_NT; nt += 4) {
        v8f acc = {};
        #pragma unroll
        for (int kt = 0; kt < WD2_KT; ++kt) {
            v16bf a = make_a_frag(arow2, kt * 32 + h * 8);
            v16bf b = load_b_frag(Wd2f, kt * WD2_NT + nt, lane);
            acc = __builtin_amdgcn_wmma_f32_16x16x32_bf16(false, a, false, b,
                                                          (short)0, acc, false, false);
        }
        int col = nt * 16 + nl;
        if (col < BB) {
            float bias = bd2[col];
            int r2 = col / BDIM, c2 = col - r2 * BDIM;
            #pragma unroll
            for (int r = 0; r < 8; ++r) {
                int m = r + 8 * h;
                int i = a0 + m;
                float v = acc[r] + bias + atom_block[(size_t)i * BB + col];
                H[(size_t)(i * BDIM + r2) * HROW + (i * BDIM + c2)] = v;
            }
        }
    }
}

extern "C" void kernel_launch(void* const* d_in, const int* in_sizes, int n_in,
                              void* d_out, int out_size, void* d_ws, size_t ws_size,
                              hipStream_t stream) {
    const float* nodes         = (const float*)d_in[0];
    const float* edges         = (const float*)d_in[1];
    const float* atom_block    = (const float*)d_in[2];
    const float* atom_in_block = (const float*)d_in[3];
    const float* Wd1 = (const float*)d_in[4];
    const float* bd1 = (const float*)d_in[5];
    const float* Wd2 = (const float*)d_in[6];
    const float* bd2 = (const float*)d_in[7];
    const float* Wo1 = (const float*)d_in[8];
    const float* bo1 = (const float*)d_in[9];
    const float* Wo2 = (const float*)d_in[10];
    const float* bo2 = (const float*)d_in[11];
    const int* pi = (const int*)d_in[12];
    const int* pj = (const int*)d_in[13];
    float* H = (float*)d_out;

    bf16* ws   = (bf16*)d_ws;   // needs ~188 KB; ws_size assumed sufficient
    bf16* Wo1f = ws + WO1_OFF;
    bf16* Wo2f = ws + WO2_OFF;
    bf16* Wd1f = ws + WD1_OFF;
    bf16* Wd2f = ws + WD2_OFF;

    // repack weights to bf16 WMMA fragments (recomputed every call: deterministic)
    {
        int t1 = WO1_KT * WO1_NT * 512;
        int t2 = WO2_KT * WO2_NT * 512;
        int t3 = WD1_KT * WD1_NT * 512;
        int t4 = WD2_KT * WD2_NT * 512;
        pack_weights_kernel<<<(t1 + 255) / 256, 256, 0, stream>>>(Wo1, Wo1f, 192, 128, WO1_NT, t1);
        pack_weights_kernel<<<(t2 + 255) / 256, 256, 0, stream>>>(Wo2, Wo2f, 128, 196, WO2_NT, t2);
        pack_weights_kernel<<<(t3 + 255) / 256, 256, 0, stream>>>(Wd1, Wd1f, 128, 128, WD1_NT, t3);
        pack_weights_kernel<<<(t4 + 255) / 256, 256, 0, stream>>>(Wd2, Wd2f, 128, 196, WD2_NT, t4);
    }

    // every H block is written exactly once by the two kernels -> no zero-fill pass
    diag_kernel<<<NATOMS / 16, 128, 0, stream>>>(nodes, edges, atom_block,
                                                 Wd1f, bd1, Wd2f, bd2, H);
    pair_kernel<<<NPAIRS / 16, 128, 0, stream>>>(nodes, edges, atom_in_block,
                                                 Wo1f, bo1, Wo2f, bo2, pi, pj, H);
}
